// EfficientCrossAttention_24180665876963
// MI455X (gfx1250) — compile-verified, matrix operand order checked by reading, not run on hardware
//
#include <hip/hip_runtime.h>

// ---------------------------------------------------------------------------
// EfficientCrossAttention for MI455X (gfx1250, wave32, WMMA bf16 + async/TDM)
// q/k/v 1x1-conv GEMMs -> dual softmax -> ctx = K V^T -> att = ctx^T q
// -> output projection -> AdaIN restyle.
// GEMMs use v_wmma_f32_16x16x32_bf16 (f32 accumulate).
// softmax_rows stages rows via global_load_async_to_lds_b128 (ASYNCcnt).
// att_kernel stages ctx via the Tensor Data Mover (TENSORcnt).
// ---------------------------------------------------------------------------

#define SDIM   16384      // H*W = 128*128
#define CCH    256        // channels (Cx = Cy = Kc = Vc)
#define NBATCH 8
#define NH     64         // n * heads
#define HCH    32         // head channels (hk = hv)

typedef __attribute__((ext_vector_type(16))) __bf16 v16bf;
typedef __attribute__((ext_vector_type(8)))  __bf16 v8bf;
typedef __attribute__((ext_vector_type(8)))  float  v8f;
typedef __attribute__((ext_vector_type(4)))  unsigned int u32x4;
typedef __attribute__((ext_vector_type(8)))  unsigned int u32x8;

// float -> bf16 (RNE), no reliance on hw cvt codegen
__device__ __forceinline__ __bf16 f2bf(float f) {
    unsigned u = __builtin_bit_cast(unsigned, f);
    u += 0x7FFFu + ((u >> 16) & 1u);
    unsigned short h = (unsigned short)(u >> 16);
    return __builtin_bit_cast(__bf16, h);
}

// two floats -> packed bf16 pair (lo = a, hi = b)
__device__ __forceinline__ unsigned pack2bf(float a, float b) {
    unsigned ua = __builtin_bit_cast(unsigned, a);
    ua += 0x7FFFu + ((ua >> 16) & 1u);
    unsigned ub = __builtin_bit_cast(unsigned, b);
    ub += 0x7FFFu + ((ub >> 16) & 1u);
    return (ua >> 16) | (ub & 0xFFFF0000u);
}

__device__ __forceinline__ v16bf combine16(v8bf lo, v8bf hi) {
    v16bf r;
#pragma unroll
    for (int i = 0; i < 8; ++i) { r[i] = lo[i]; r[8 + i] = hi[i]; }
    return r;
}

// 8 consecutive f32 from global -> v8bf (two b128 loads)
__device__ __forceinline__ v8bf load8f_bf(const float* __restrict__ p) {
    float4 a = *(const float4*)p;
    float4 b = *(const float4*)(p + 4);
    v8bf r;
    r[0] = f2bf(a.x); r[1] = f2bf(a.y); r[2] = f2bf(a.z); r[3] = f2bf(a.w);
    r[4] = f2bf(b.x); r[5] = f2bf(b.y); r[6] = f2bf(b.z); r[7] = f2bf(b.w);
    return r;
}

__device__ __forceinline__ v8f wmma_bf16(v16bf a, v16bf b, v8f c) {
    return __builtin_amdgcn_wmma_f32_16x16x32_bf16(
        /*neg_a=*/false, a, /*neg_b=*/false, b,
        /*c_mod=*/(short)0, c, /*reuse_a=*/false, /*reuse_b=*/false);
}

// generic (shared) pointer -> 32-bit LDS byte offset
__device__ __forceinline__ unsigned lds_off32(const void* p) {
    return (unsigned)(unsigned long long)(uintptr_t)p;
}

// ---------------------------------------------------------------------------
// Kernel 1: OUT[n][m][s] = sum_k W[m][k] * IN[n][k][s] + bias[m]
// Block tile 128(m) x 64(s); 8 waves in 4x2; each wave 32x32 (2x2 WMMA).
// grid = (SDIM/64, CCH/128, NBATCH)
// ---------------------------------------------------------------------------
__global__ __launch_bounds__(256)
void proj_gemm(const float* __restrict__ W, const float* __restrict__ bias,
               const float* __restrict__ IN, float* __restrict__ OUT) {
    __shared__ __bf16 ldsA[128][32];   // weights  [m][k]
    __shared__ __bf16 ldsB[64][32];    // input^T  [s][k]

    const int tid  = threadIdx.x;
    const int wave = tid >> 5;
    const int lane = tid & 31;
    const int half = lane >> 4;
    const int l16  = lane & 15;

    const int n    = blockIdx.z;
    const int mBlk = blockIdx.y * 128;
    const int sBlk = blockIdx.x * 64;

    const float* __restrict__ inN  = IN  + (size_t)n * CCH * SDIM;
    float* __restrict__       outN = OUT + (size_t)n * CCH * SDIM;

    unsigned* __restrict__ ldsAu = (unsigned*)&ldsA[0][0];
    unsigned* __restrict__ ldsBu = (unsigned*)&ldsB[0][0];

    const int wm = (wave & 3) * 32;   // wave m offset inside the 128 tile
    const int ws = (wave >> 2) * 32;  // wave s offset inside the 64 tile

    v8f acc[2][2] = {};

    for (int kk = 0; kk < CCH; kk += 32) {
        // stage W tile 128x32 as packed bf16 pairs (row-major [m][k])
        for (int i = tid; i < 128 * 16; i += 256) {
            const int m = i >> 4, kp = (i & 15) * 2;
            const float* wp = &W[(size_t)(mBlk + m) * CCH + kk + kp];
            ldsAu[i] = pack2bf(wp[0], wp[1]);
        }
        // stage IN tile 32x64 transposed into [s][k], packed bf16 pairs
        for (int i = tid; i < 16 * 64; i += 256) {
            const int kp = i >> 6, s = i & 63;   // kp: pair index 0..15
            const float f0 = inN[(size_t)(kk + 2 * kp) * SDIM + sBlk + s];
            const float f1 = inN[(size_t)(kk + 2 * kp + 1) * SDIM + sBlk + s];
            ldsBu[s * 16 + kp] = pack2bf(f0, f1);
        }
        if (kk + 32 < CCH)  // hint next input tile toward L2
            __builtin_prefetch(&inN[(size_t)(kk + 32 + (tid >> 6)) * SDIM + sBlk], 0, 1);
        __syncthreads();

        v16bf afrag[2], bfrag[2];
#pragma unroll
        for (int mt = 0; mt < 2; ++mt) {
            const __bf16* row = &ldsA[wm + mt * 16 + l16][0];
            afrag[mt] = combine16(*(const v8bf*)(row + half * 8),
                                  *(const v8bf*)(row + 16 + half * 8));
        }
#pragma unroll
        for (int nt = 0; nt < 2; ++nt) {
            const __bf16* row = &ldsB[ws + nt * 16 + l16][0];
            bfrag[nt] = combine16(*(const v8bf*)(row + half * 8),
                                  *(const v8bf*)(row + 16 + half * 8));
        }
#pragma unroll
        for (int mt = 0; mt < 2; ++mt)
#pragma unroll
            for (int nt = 0; nt < 2; ++nt)
                acc[mt][nt] = wmma_bf16(afrag[mt], bfrag[nt], acc[mt][nt]);
        __syncthreads();
    }

    // C layout: lane l, vgpr i -> M = i + 8*(l>=16), N = l%16
#pragma unroll
    for (int mt = 0; mt < 2; ++mt) {
#pragma unroll
        for (int nt = 0; nt < 2; ++nt) {
            const int col = sBlk + ws + nt * 16 + l16;
#pragma unroll
            for (int i = 0; i < 8; ++i) {
                const int row = mBlk + wm + mt * 16 + 8 * half + i;
                outN[(size_t)row * SDIM + col] = acc[mt][nt][i] + bias[row];
            }
        }
    }
}

// ---------------------------------------------------------------------------
// Kernel 2: softmax over the sequence axis for K. One block per (n, channel).
// The 64KB row is staged into LDS once via async global->LDS copies; the
// three passes (max / sum / normalize) then read LDS instead of re-reading HBM.
// ---------------------------------------------------------------------------
__global__ __launch_bounds__(256)
void softmax_rows(float* __restrict__ K) {
    __shared__ float rowbuf[SDIM];     // 64KB of the 320KB WGP LDS
    __shared__ float red[256];
    const int tid = threadIdx.x;
    float* __restrict__ p = K + (size_t)blockIdx.x * SDIM;

    // async stage: 16B per lane per op, 16 ops per thread
    const unsigned lbase = lds_off32(&rowbuf[0]);
    for (int i = tid * 4; i < SDIM; i += 256 * 4) {
        const unsigned la = lbase + (unsigned)i * 4u;
        const unsigned long long ga = (unsigned long long)(uintptr_t)(p + i);
        asm volatile("global_load_async_to_lds_b128 %0, %1, off"
                     :: "v"(la), "v"(ga) : "memory");
    }
    asm volatile("s_wait_asynccnt 0x0" ::: "memory");
    __syncthreads();

    float m = -3.402823466e38f;
    for (int s = tid; s < SDIM; s += 256) m = fmaxf(m, rowbuf[s]);
    red[tid] = m; __syncthreads();
    for (int off = 128; off; off >>= 1) {
        if (tid < off) red[tid] = fmaxf(red[tid], red[tid + off]);
        __syncthreads();
    }
    const float rowmax = red[0]; __syncthreads();

    float sum = 0.f;
    for (int s = tid; s < SDIM; s += 256) sum += __expf(rowbuf[s] - rowmax);
    red[tid] = sum; __syncthreads();
    for (int off = 128; off; off >>= 1) {
        if (tid < off) red[tid] += red[tid + off];
        __syncthreads();
    }
    const float inv = 1.0f / red[0];

    for (int s = tid; s < SDIM; s += 256)
        p[s] = __expf(rowbuf[s] - rowmax) * inv;
}

// ---------------------------------------------------------------------------
// Kernel 3: softmax over the 32 head-channels of Q, per pixel.
// grid = (SDIM/256, heads, n)
// ---------------------------------------------------------------------------
__global__ __launch_bounds__(256)
void softmax_q(float* __restrict__ Q) {
    const int s = blockIdx.x * 256 + threadIdx.x;
    const int h = blockIdx.y, n = blockIdx.z;
    float* __restrict__ base = Q + (size_t)(n * CCH + h * HCH) * SDIM + s;

    float v[HCH];
    float m = -3.402823466e38f;
#pragma unroll
    for (int j = 0; j < HCH; ++j) {
        v[j] = base[(size_t)j * SDIM];
        m = fmaxf(m, v[j]);
    }
    float sum = 0.f;
#pragma unroll
    for (int j = 0; j < HCH; ++j) { v[j] = __expf(v[j] - m); sum += v[j]; }
    const float inv = 1.0f / sum;
#pragma unroll
    for (int j = 0; j < HCH; ++j) base[(size_t)j * SDIM] = v[j] * inv;
}

// ---------------------------------------------------------------------------
// Kernel 4: ctx[nh][hk][hv] = sum_s K[nh][hk][s] * V[nh][hv][s]
// One block per (n,h); each wave strip-mines s with 2x2 WMMA, LDS f32 reduce.
// ---------------------------------------------------------------------------
__global__ __launch_bounds__(256)
void ctx_kernel(const float* __restrict__ K, const float* __restrict__ V,
                float* __restrict__ CTX) {
    __shared__ float red[HCH * HCH];
    const int tid  = threadIdx.x;
    const int wave = tid >> 5;
    const int lane = tid & 31;
    const int half = lane >> 4;
    const int l16  = lane & 15;
    const int nh   = blockIdx.x;

    const float* __restrict__ kb = K + (size_t)nh * HCH * SDIM;
    const float* __restrict__ vb = V + (size_t)nh * HCH * SDIM;

    for (int i = tid; i < HCH * HCH; i += 256) red[i] = 0.f;
    __syncthreads();

    v8f acc[2][2] = {};
    for (int s = wave * 32; s < SDIM; s += 256) {
        v16bf af[2], bf[2];
#pragma unroll
        for (int mt = 0; mt < 2; ++mt) {   // A[hk][s-chunk]
            const float* rp = kb + (size_t)(mt * 16 + l16) * SDIM + s;
            af[mt] = combine16(load8f_bf(rp + half * 8),
                               load8f_bf(rp + 16 + half * 8));
        }
#pragma unroll
        for (int nt = 0; nt < 2; ++nt) {   // B[s-chunk][hv] = V[hv][s]
            const float* cp = vb + (size_t)(nt * 16 + l16) * SDIM + s;
            bf[nt] = combine16(load8f_bf(cp + half * 8),
                               load8f_bf(cp + 16 + half * 8));
        }
#pragma unroll
        for (int mt = 0; mt < 2; ++mt)
#pragma unroll
            for (int nt = 0; nt < 2; ++nt)
                acc[mt][nt] = wmma_bf16(af[mt], bf[nt], acc[mt][nt]);
    }

#pragma unroll
    for (int mt = 0; mt < 2; ++mt)
#pragma unroll
        for (int nt = 0; nt < 2; ++nt)
#pragma unroll
            for (int i = 0; i < 8; ++i) {
                const int r = mt * 16 + 8 * half + i;
                const int c = nt * 16 + l16;
                atomicAdd(&red[r * HCH + c], acc[mt][nt][i]);   // ds_add_f32
            }
    __syncthreads();

    float* __restrict__ out = CTX + (size_t)nh * HCH * HCH;
    for (int i = tid; i < HCH * HCH; i += 256) out[i] = red[i];
}

// ---------------------------------------------------------------------------
// Kernel 5: att[nh][hv][s] = sum_hk ctx[nh][hk][hv] * q[nh][hk][s]
// grid = (SDIM/128, NH). The 4KB ctx block is DMA'd into LDS by the Tensor
// Data Mover (tensor_load_to_lds + s_wait_tensorcnt); fragments are built
// from the raw f32 LDS copy with on-the-fly bf16 conversion. K=32 -> 1 WMMA.
// ---------------------------------------------------------------------------
__global__ __launch_bounds__(256)
void att_kernel(const float* __restrict__ CTX, const float* __restrict__ Q,
                float* __restrict__ ATT) {
    __shared__ float ldsCtx[HCH * HCH];   // [hk][hv], straight copy of ctx
    const int tid  = threadIdx.x;
    const int wave = tid >> 5;
    const int lane = tid & 31;
    const int half = lane >> 4;
    const int l16  = lane & 15;
    const int nh   = blockIdx.y;

    const float* __restrict__ ctx = CTX + (size_t)nh * HCH * HCH;

    if (wave == 0) {
        // Tensor DMA descriptor: 1D tile of 1024 x 4B elements, global -> LDS.
        const unsigned long long ga = (unsigned long long)(uintptr_t)ctx;
        u32x4 g0;
        g0[0] = 1u;                                   // count=1, user mode
        g0[1] = lds_off32(&ldsCtx[0]);                // lds_addr
        g0[2] = (unsigned)ga;                         // global_addr[31:0]
        g0[3] = (unsigned)((ga >> 32) & 0x01FFFFFFull) | (2u << 30); // addr hi | type=2
        u32x8 g1;
        g1[0] = (2u << 16);        // workgroup_mask=0, data_size=2 (4 bytes)
        g1[1] = (1024u << 16);     // tensor_dim0 = 1024 (low 16 in [31:16])
        g1[2] = (1u << 16);        // tensor_dim0 hi=0, tensor_dim1 = 1
        g1[3] = (1024u << 16);     // tensor_dim1 hi=0, tile_dim0 = 1024
        g1[4] = 1u;                // tile_dim1 = 1, tile_dim2 = 0
        g1[5] = 1024u;             // tensor_dim0_stride lo
        g1[6] = 0u;                // stride0 hi, stride1 lo
        g1[7] = 0u;                // stride1 hi
        asm volatile("tensor_load_to_lds %0, %1" :: "s"(g0), "s"(g1) : "memory");
        __builtin_amdgcn_s_wait_tensorcnt(0);
    }
    __syncthreads();

    const size_t rowbase = (size_t)((nh >> 3) * CCH + (nh & 7) * HCH) * SDIM;
    const float* __restrict__ qb = Q + rowbase;
    float* __restrict__       ab = ATT + rowbase;

    const int s = blockIdx.x * 128 + wave * 16 + l16;

    v16bf bfrag;   // B[hk][s]: strided gather down the channel axis
#pragma unroll
    for (int j = 0; j < 8; ++j) {
        bfrag[j]     = f2bf(qb[(size_t)(half * 8 + j) * SDIM + s]);
        bfrag[8 + j] = f2bf(qb[(size_t)(16 + half * 8 + j) * SDIM + s]);
    }

#pragma unroll
    for (int mt = 0; mt < 2; ++mt) {
        const int hv = mt * 16 + l16;
        v16bf af;   // A[hv][hk] = ctx[hk][hv]: transposed reads from f32 LDS
#pragma unroll
        for (int j = 0; j < 8; ++j) {
            af[j]     = f2bf(ldsCtx[(half * 8 + j) * HCH + hv]);
            af[8 + j] = f2bf(ldsCtx[(16 + half * 8 + j) * HCH + hv]);
        }
        v8f acc = {};
        acc = wmma_bf16(af, bfrag, acc);
#pragma unroll
        for (int i = 0; i < 8; ++i) {
            const int hvr = mt * 16 + 8 * half + i;
            ab[(size_t)hvr * SDIM + s] = acc[i];
        }
    }
}

// ---------------------------------------------------------------------------
// Kernel 6: per-(n,c) instance stats of x and attn.
// st layout: [0..2047]=mean_x, [2048..]=rstd_x, [4096..]=mean_a, [6144..]=std_a
// ---------------------------------------------------------------------------
__global__ __launch_bounds__(256)
void stats_kernel(const float* __restrict__ X, const float* __restrict__ A,
                  float* __restrict__ st) {
    __shared__ float r0[256], r1[256], r2[256], r3[256];
    const int row = blockIdx.x, tid = threadIdx.x;
    const float* __restrict__ xp = X + (size_t)row * SDIM;
    const float* __restrict__ ap = A + (size_t)row * SDIM;

    float sx = 0.f, sxx = 0.f, sa = 0.f, saa = 0.f;
    for (int s = tid; s < SDIM; s += 256) {
        const float x = xp[s], a = ap[s];
        sx += x; sxx += x * x; sa += a; saa += a * a;
    }
    r0[tid] = sx; r1[tid] = sxx; r2[tid] = sa; r3[tid] = saa;
    __syncthreads();
    for (int off = 128; off; off >>= 1) {
        if (tid < off) {
            r0[tid] += r0[tid + off]; r1[tid] += r1[tid + off];
            r2[tid] += r2[tid + off]; r3[tid] += r3[tid + off];
        }
        __syncthreads();
    }
    if (tid == 0) {
        const float inv = 1.0f / (float)SDIM;
        const float mx = r0[0] * inv, vx = r1[0] * inv - mx * mx;
        const float ma = r2[0] * inv, va = r3[0] * inv - ma * ma;
        st[row]            = mx;
        st[2048 + row]     = rsqrtf(vx + 1e-5f);
        st[2 * 2048 + row] = ma;
        st[3 * 2048 + row] = sqrtf(va + 1e-5f);
    }
}

// ---------------------------------------------------------------------------
// Kernel 7: AdaIN restyle: out = std_a * (x - mx) * rstd_x + mean_a
// ---------------------------------------------------------------------------
__global__ __launch_bounds__(256)
void adain_kernel(const float* __restrict__ X, const float* __restrict__ st,
                  float* __restrict__ OUT) {
    const int row = blockIdx.x, tid = threadIdx.x;
    const float mx  = st[row];
    const float rsx = st[2048 + row];
    const float ma  = st[2 * 2048 + row];
    const float sa  = st[3 * 2048 + row];
    const float scale = sa * rsx;
    const float shift = ma - scale * mx;

    const float4* __restrict__ xp = (const float4*)(X + (size_t)row * SDIM);
    float4* __restrict__       op = (float4*)(OUT + (size_t)row * SDIM);
    for (int i = tid; i < SDIM / 4; i += 256) {
        const float4 x = xp[i];
        float4 o;
        o.x = x.x * scale + shift; o.y = x.y * scale + shift;
        o.z = x.z * scale + shift; o.w = x.w * scale + shift;
        op[i] = o;
    }
}

// ---------------------------------------------------------------------------
extern "C" void kernel_launch(void* const* d_in, const int* in_sizes, int n_in,
                              void* d_out, int out_size, void* d_ws, size_t ws_size,
                              hipStream_t stream) {
    const float* x  = (const float*)d_in[0];
    const float* y  = (const float*)d_in[1];
    const float* Wq = (const float*)d_in[2];
    const float* bq = (const float*)d_in[3];
    const float* Wk = (const float*)d_in[4];
    const float* bk = (const float*)d_in[5];
    const float* Wv = (const float*)d_in[6];
    const float* bv = (const float*)d_in[7];
    const float* Wr = (const float*)d_in[8];
    const float* br = (const float*)d_in[9];
    float* out = (float*)d_out;

    float* ws = (float*)d_ws;
    const size_t NQ = (size_t)NBATCH * CCH * SDIM;   // 33.5M floats per tensor
    float* q    = ws;
    float* k    = ws + NQ;
    float* v    = ws + 2 * NQ;
    float* ctxb = ws + 3 * NQ;                        // 64 * 32 * 32
    float* st   = ctxb + (size_t)NH * HCH * HCH;      // 4 * 2048
    float* att  = k;   // k is dead after ctx_kernel
    float* attn = v;   // v is dead after ctx_kernel

    const dim3 blk(256);
    const dim3 gProj(SDIM / 64, CCH / 128, NBATCH);

    proj_gemm<<<gProj, blk, 0, stream>>>(Wq, bq, x, q);
    proj_gemm<<<gProj, blk, 0, stream>>>(Wk, bk, y, k);
    proj_gemm<<<gProj, blk, 0, stream>>>(Wv, bv, y, v);

    softmax_rows<<<dim3(NBATCH * CCH), blk, 0, stream>>>(k);
    softmax_q<<<dim3(SDIM / 256, 8, NBATCH), blk, 0, stream>>>(q);

    ctx_kernel<<<dim3(NH), blk, 0, stream>>>(k, v, ctxb);
    att_kernel<<<dim3(SDIM / 128, NH), blk, 0, stream>>>(ctxb, q, att);

    proj_gemm<<<gProj, blk, 0, stream>>>(Wr, br, att, attn);

    stats_kernel<<<dim3(NBATCH * CCH), blk, 0, stream>>>(x, attn, st);
    adain_kernel<<<dim3(NBATCH * CCH), blk, 0, stream>>>(x, st, out);
}
